// TGAMultiHeadAttion_90168543412703
// MI455X (gfx1250) — matrix operand order repaired and checked by reading, not compiled, not run on hardware
//
#include <hip/hip_runtime.h>
#include <hip/hip_bf16.h>
#include <stdint.h>

typedef __attribute__((ext_vector_type(16))) _Float16 v16h;
typedef __attribute__((ext_vector_type(8)))  _Float16 v8h;
typedef __attribute__((ext_vector_type(4)))  _Float16 v4h;
typedef __attribute__((ext_vector_type(8)))  float    v8f;
typedef __attribute__((ext_vector_type(4)))  float    f32x4;
typedef __attribute__((ext_vector_type(4)))  unsigned u32x4;
typedef __attribute__((ext_vector_type(4)))  int      i32x4;
typedef __attribute__((ext_vector_type(8)))  int      i32x8;

#define B_  8
#define S_  1024
#define D_  512
#define H_  8
#define HD_ 64
#define M_  (B_ * S_)   // 8192

#if __has_builtin(__builtin_amdgcn_tensor_load_to_lds) && __has_builtin(__builtin_amdgcn_s_wait_tensorcnt)
#define USE_TDM 1
#else
#define USE_TDM 0
#endif

static __device__ __forceinline__ v8f wmma16(const v16h& a, const v16h& b, const v8f& c) {
    return __builtin_amdgcn_wmma_f32_16x16x32_f16(false, a, false, b, (short)0, c, false, false);
}

#if USE_TDM
// TDM: DMA a 64-row x 64-half tile (global row stride 512 halves) into LDS,
// padding the LDS destination by 4 DWORDs (8 halves) after every 32 DWORDs
// (128B row) -> LDS row stride 72 halves, matching the Kt/Qt layouts.
static __device__ __forceinline__ void tdm_load_tile64(const _Float16* gsrc, void* ldst) {
    uint64_t ga = (uint64_t)(uintptr_t)gsrc;
    u32x4 g0;
    g0[0] = 1u;                                   // count=1 (valid user descriptor)
    g0[1] = (unsigned)(uintptr_t)ldst;            // lds_addr (low 32 bits of flat = LDS offset)
    g0[2] = (unsigned)ga;                         // global_addr[31:0]
    g0[3] = (unsigned)((ga >> 32) & 0x01FFFFFFu)  // global_addr[56:32]
          | (2u << 30);                           // type=2 ("image")
    i32x8 g1;
    g1[0] = (1 << 16)        // data_size = 1 -> 2-byte elements
          | (1 << 20)        // pad_enable
          | (4 << 22)        // pad_interval: pad after 32 DWORDs (128B)
          | (3 << 25);       // pad_amount: 4 DWORDs (16B = 8 halves)
    g1[1] = (64 << 16);      // tensor_dim0[15:0] = 64 (abar addr = 0)
    g1[2] = (64 << 16);      // tensor_dim0[31:16]=0 ; tensor_dim1[15:0] = 64
    g1[3] = (64 << 16);      // tensor_dim1[31:16]=0 ; tile_dim0 = 64
    g1[4] = 64;              // tile_dim1 = 64 ; tile_dim2 = 0
    g1[5] = 512;             // tensor_dim0_stride = 512 elements
    g1[6] = 0;
    g1[7] = 0;
    i32x4 g2 = {0, 0, 0, 0};
    i32x4 g3 = {0, 0, 0, 0};
#if defined(__clang_major__) && (__clang_major__ >= 23)
    i32x8 g4 = {0, 0, 0, 0, 0, 0, 0, 0};
    __builtin_amdgcn_tensor_load_to_lds(g0, g1, g2, g3, g4, 0);
#else
    __builtin_amdgcn_tensor_load_to_lds(g0, g1, g2, g3, 0);
#endif
}
#endif

// ---------------------------------------------------------------- fp32 -> fp16
__global__ __launch_bounds__(256) void cvt_f32_f16_kernel(const float* __restrict__ src,
                                                          _Float16* __restrict__ dst, int n) {
    int i = (blockIdx.x * 256 + threadIdx.x) * 4;
    if (i + 3 < n) {
        float4 v = *(const float4*)(src + i);
        v4h o;
        o[0] = (_Float16)v.x; o[1] = (_Float16)v.y; o[2] = (_Float16)v.z; o[3] = (_Float16)v.w;
        *(v4h*)(dst + i) = o;
    }
}

// ----------------------------------------------- fused GEMM (x @ W^T) + LayerNorm
// grid: (M_/32, 1, 3)  block: 256 (8 waves). Each block: 32 rows x all 512 cols.
__global__ __launch_bounds__(256)
void gemm_ln_kernel(const _Float16* __restrict__ xh,   // [8192][512]
                    const _Float16* __restrict__ wh,   // [3][512][512] (row n, contiguous d)
                    const float* __restrict__ bq, const float* __restrict__ bk,
                    const float* __restrict__ bv,
                    const float* __restrict__ gamma, const float* __restrict__ beta,
                    _Float16* __restrict__ qkvh)       // [3][8192][512]
{
    const int z     = blockIdx.z;
    const int mbase = blockIdx.x * 32;
    const _Float16* W    = wh + (size_t)z * D_ * D_;
    const float*    bias = (z == 0) ? bq : (z == 1) ? bk : bv;
    _Float16*       Yout = qkvh + (size_t)z * M_ * D_;

    __shared__ _Float16 At[32 * 40];    // padded stride 40 halves (80B, 16B-aligned rows)
    __shared__ _Float16 Wt[512 * 40];
    __shared__ float    s_sum[32];
    __shared__ float    s_sq[32];

    const int t    = threadIdx.x;
    const int wv   = t >> 5;
    const int lane = t & 31;
    const int l15  = lane & 15;
    const int hi   = lane >> 4;          // 0 or 1
    const int rt   = wv & 1;             // row tile (16 rows)
    const int ct0  = (wv >> 1) * 8;      // 8 col tiles = 128 cols per wave

    v8f acc[8];
    #pragma unroll
    for (int i = 0; i < 8; ++i)
        #pragma unroll
        for (int j = 0; j < 8; ++j) acc[i][j] = 0.f;

    for (int k0 = 0; k0 < D_; k0 += 32) {
        { // stage A: 32 rows x 32 halves
            int row = t >> 3, kk = (t & 7) * 4;
            *(v4h*)&At[row * 40 + kk] = *(const v4h*)&xh[(size_t)(mbase + row) * D_ + k0 + kk];
        }
        { // stage W: 512 rows x 32 halves (2 rows / thread)
            int n0 = t * 2;
            #pragma unroll
            for (int r = 0; r < 2; ++r) {
                int n = n0 + r;
                const _Float16* src = &W[(size_t)n * D_ + k0];
                _Float16* dst = &Wt[n * 40];
                #pragma unroll
                for (int i = 0; i < 4; ++i)
                    *(v8h*)&dst[i * 8] = *(const v8h*)&src[i * 8];
            }
        }
        if (k0 + 32 < D_) { // prefetch next K-step (global_prefetch_b8)
            __builtin_prefetch(&xh[(size_t)(mbase + (t >> 3)) * D_ + k0 + 32], 0, 0);
            __builtin_prefetch(&W[(size_t)(t * 2) * D_ + k0 + 32], 0, 0);
        }
        __syncthreads();

        v16h a; // A fragment 16x32 for this wave's row tile
        {
            int row = rt * 16 + l15;
            int kb  = hi ? 8 : 0;
            v8h lo = *(const v8h*)&At[row * 40 + kb];
            v8h hh = *(const v8h*)&At[row * 40 + kb + 16];
            #pragma unroll
            for (int i = 0; i < 8; ++i) { a[i] = lo[i]; a[8 + i] = hh[i]; }
        }
        #pragma unroll
        for (int tile = 0; tile < 8; ++tile) { // B fragments 32x16
            int n  = (ct0 + tile) * 16 + l15;
            int kb = hi ? 8 : 0;
            v8h b0 = *(const v8h*)&Wt[n * 40 + kb];
            v8h b1 = *(const v8h*)&Wt[n * 40 + kb + 16];
            v16h bf;
            #pragma unroll
            for (int i = 0; i < 8; ++i) { bf[i] = b0[i]; bf[8 + i] = b1[i]; }
            acc[tile] = wmma16(a, bf, acc[tile]);
        }
        __syncthreads();
    }

    // bias
    #pragma unroll
    for (int tile = 0; tile < 8; ++tile) {
        int n = (ct0 + tile) * 16 + l15;
        float bb = bias[n];
        #pragma unroll
        for (int j = 0; j < 8; ++j) acc[tile][j] += bb;
    }

    // LayerNorm row stats: tile-sum -> 16-lane shfl reduce -> LDS atomics
    if (t < 32) { s_sum[t] = 0.f; s_sq[t] = 0.f; }
    __syncthreads();
    v8f psum, psq;
    #pragma unroll
    for (int j = 0; j < 8; ++j) { psum[j] = 0.f; psq[j] = 0.f; }
    #pragma unroll
    for (int tile = 0; tile < 8; ++tile) {
        psum += acc[tile];
        psq  += acc[tile] * acc[tile];
    }
    #pragma unroll
    for (int m = 1; m < 16; m <<= 1)
        #pragma unroll
        for (int j = 0; j < 8; ++j) {
            psum[j] += __shfl_xor(psum[j], m, 32);
            psq[j]  += __shfl_xor(psq[j],  m, 32);
        }
    if (l15 == 0) {
        int rbase = rt * 16 + (hi ? 8 : 0);
        #pragma unroll
        for (int j = 0; j < 8; ++j) {
            atomicAdd(&s_sum[rbase + j], psum[j]);
            atomicAdd(&s_sq[rbase + j],  psq[j]);
        }
    }
    __syncthreads();

    // normalize + store f16
    #pragma unroll
    for (int tile = 0; tile < 8; ++tile) {
        int n = (ct0 + tile) * 16 + l15;
        float g = gamma[n], bt = beta[n];
        #pragma unroll
        for (int j = 0; j < 8; ++j) {
            int rloc  = rt * 16 + j + (hi ? 8 : 0);
            float mean = s_sum[rloc] * (1.0f / D_);
            float var  = s_sq[rloc] * (1.0f / D_) - mean * mean;
            float rstd = rsqrtf(var + 1e-5f);
            float y = (acc[tile][j] - mean) * rstd * g + bt;
            Yout[(size_t)(mbase + rloc) * D_ + n] = (_Float16)y;
        }
    }
}

// ------------------------------------ flash-style negative-distance attention
// grid: (S/64, H, B)  block: 128 (4 waves); wave w owns query rows w*16..w*16+15.
__global__ __launch_bounds__(128)
void attn_kernel(const _Float16* __restrict__ qkvh, float* __restrict__ out)
{
    const int qt = blockIdx.x, h = blockIdx.y, b = blockIdx.z;
    const _Float16* Qh = qkvh;
    const _Float16* Kh = qkvh + (size_t)M_ * D_;
    const _Float16* Vh = qkvh + (size_t)2 * M_ * D_;

    __shared__ _Float16 Qt[64 * 72];
    __shared__ _Float16 Kt[64 * 72];
    __shared__ _Float16 Vt[64 * 72];   // transposed: [d][key]
    __shared__ _Float16 Pt[64 * 72];
    __shared__ float s_q2[64];
    __shared__ float s_k2[64];

    const int t    = threadIdx.x;
    const int wv   = t >> 5;
    const int lane = t & 31;
    const int l15  = lane & 15;
    const int hi   = lane >> 4;

    const size_t rowQ0 = (size_t)b * S_ + qt * 64;
    const int    colH  = h * HD_;

#if USE_TDM
    if (wv == 0) {
        tdm_load_tile64(&Qh[rowQ0 * D_ + colH], &Qt[0]);
        __builtin_amdgcn_s_wait_tensorcnt(0);
    }
#else
    { // stage Q tile 64x64
        int row = t >> 1, db = (t & 1) * 32;
        const _Float16* src = &Qh[(rowQ0 + row) * D_ + colH + db];
        _Float16* dst = &Qt[row * 72 + db];
        #pragma unroll
        for (int i = 0; i < 4; ++i) *(v8h*)&dst[i * 8] = *(const v8h*)&src[i * 8];
    }
#endif
    __syncthreads();
    if (t < 64) {
        float s = 0.f;
        for (int d = 0; d < 64; ++d) { float q = (float)Qt[t * 72 + d]; s += q * q; }
        s_q2[t] = s;
    }
    __syncthreads();

    v16h aq[2]; // Q A-fragments (16 rows x 64 K, two K=32 chunks)
    {
        int row = wv * 16 + l15;
        int kb  = hi ? 8 : 0;
        #pragma unroll
        for (int c = 0; c < 2; ++c) {
            v8h lo = *(const v8h*)&Qt[row * 72 + c * 32 + kb];
            v8h hh = *(const v8h*)&Qt[row * 72 + c * 32 + kb + 16];
            #pragma unroll
            for (int i = 0; i < 8; ++i) { aq[c][i] = lo[i]; aq[c][8 + i] = hh[i]; }
        }
    }

    float mrow[8], lrow[8], myq2[8];
    v8f   oacc[4];
    {
        int rbase = wv * 16 + (hi ? 8 : 0);
        #pragma unroll
        for (int j = 0; j < 8; ++j) { mrow[j] = -3.0e38f; lrow[j] = 0.f; myq2[j] = s_q2[rbase + j]; }
        #pragma unroll
        for (int i = 0; i < 4; ++i)
            #pragma unroll
            for (int j = 0; j < 8; ++j) oacc[i][j] = 0.f;
    }

    for (int kb0 = 0; kb0 < S_; kb0 += 64) {
        __syncthreads(); // previous step's Kt/Vt/Pt consumers done
#if USE_TDM
        if (wv == 0) // async DMA of K tile overlaps the manual V transpose below
            tdm_load_tile64(&Kh[((size_t)b * S_ + kb0) * D_ + colH], &Kt[0]);
#endif
        { // stage V transposed
            int key = t >> 1, db = (t & 1) * 32;
            const _Float16* vsrc = &Vh[((size_t)b * S_ + kb0 + key) * D_ + colH + db];
            #pragma unroll
            for (int i = 0; i < 32; ++i) Vt[(db + i) * 72 + key] = vsrc[i];
        }
#if !USE_TDM
        { // stage K row-major
            int key = t >> 1, db = (t & 1) * 32;
            const _Float16* ksrc = &Kh[((size_t)b * S_ + kb0 + key) * D_ + colH + db];
            _Float16* kdst = &Kt[key * 72 + db];
            #pragma unroll
            for (int i = 0; i < 4; ++i) *(v8h*)&kdst[i * 8] = *(const v8h*)&ksrc[i * 8];
        }
#else
        if (wv == 0) __builtin_amdgcn_s_wait_tensorcnt(0);
#endif
        __syncthreads();
        if (t < 64) {
            float s = 0.f;
            for (int d = 0; d < 64; ++d) { float k = (float)Kt[t * 72 + d]; s += k * k; }
            s_k2[t] = s;
        }
        __syncthreads();

        // scores: -sqrt(q2 + k2 - 2 Q.K)
        v8f sc[4];
        #pragma unroll
        for (int kt = 0; kt < 4; ++kt) {
            v8f dot;
            #pragma unroll
            for (int j = 0; j < 8; ++j) dot[j] = 0.f;
            int key = kt * 16 + l15;
            #pragma unroll
            for (int c = 0; c < 2; ++c) {
                int dbb = c * 32 + (hi ? 8 : 0);
                v8h b0 = *(const v8h*)&Kt[key * 72 + dbb];
                v8h b1 = *(const v8h*)&Kt[key * 72 + dbb + 16];
                v16h bf;
                #pragma unroll
                for (int i = 0; i < 8; ++i) { bf[i] = b0[i]; bf[8 + i] = b1[i]; }
                dot = wmma16(aq[c], bf, dot);
            }
            float k2c = s_k2[key];
            #pragma unroll
            for (int j = 0; j < 8; ++j) {
                float d2 = fmaxf(myq2[j] + k2c - 2.0f * dot[j], 1e-12f);
                sc[kt][j] = -sqrtf(d2);
            }
        }

        // online softmax (rows = accumulator VGPR index; cols = lanes)
        v8f tmax = sc[0];
        #pragma unroll
        for (int kt = 1; kt < 4; ++kt)
            #pragma unroll
            for (int j = 0; j < 8; ++j) tmax[j] = fmaxf(tmax[j], sc[kt][j]);
        #pragma unroll
        for (int m = 1; m < 16; m <<= 1)
            #pragma unroll
            for (int j = 0; j < 8; ++j) tmax[j] = fmaxf(tmax[j], __shfl_xor(tmax[j], m, 32));
        float corr[8], mnew[8];
        #pragma unroll
        for (int j = 0; j < 8; ++j) {
            mnew[j] = fmaxf(mrow[j], tmax[j]);
            corr[j] = __expf(mrow[j] - mnew[j]);
            mrow[j] = mnew[j];
        }
        v8f psum;
        #pragma unroll
        for (int j = 0; j < 8; ++j) psum[j] = 0.f;
        #pragma unroll
        for (int kt = 0; kt < 4; ++kt) {
            int col = kt * 16 + l15;
            #pragma unroll
            for (int j = 0; j < 8; ++j) {
                float p = __expf(sc[kt][j] - mnew[j]);
                psum[j] += p;
                int row = wv * 16 + j + (hi ? 8 : 0);
                Pt[row * 72 + col] = (_Float16)p;
            }
        }
        #pragma unroll
        for (int m = 1; m < 16; m <<= 1)
            #pragma unroll
            for (int j = 0; j < 8; ++j) psum[j] += __shfl_xor(psum[j], m, 32);
        #pragma unroll
        for (int j = 0; j < 8; ++j) lrow[j] = lrow[j] * corr[j] + psum[j];
        #pragma unroll
        for (int nt = 0; nt < 4; ++nt)
            #pragma unroll
            for (int j = 0; j < 8; ++j) oacc[nt][j] *= corr[j];
        __syncthreads(); // Pt visible

        // P A-fragments, then P.V
        v16h ap[2];
        {
            int row = wv * 16 + l15;
            int kb  = hi ? 8 : 0;
            #pragma unroll
            for (int c = 0; c < 2; ++c) {
                v8h lo = *(const v8h*)&Pt[row * 72 + c * 32 + kb];
                v8h hh = *(const v8h*)&Pt[row * 72 + c * 32 + kb + 16];
                #pragma unroll
                for (int i = 0; i < 8; ++i) { ap[c][i] = lo[i]; ap[c][8 + i] = hh[i]; }
            }
        }
        #pragma unroll
        for (int nt = 0; nt < 4; ++nt) {
            int dcol = nt * 16 + l15;
            #pragma unroll
            for (int c = 0; c < 2; ++c) {
                int kk = c * 32 + (hi ? 8 : 0);
                v8h b0 = *(const v8h*)&Vt[dcol * 72 + kk];
                v8h b1 = *(const v8h*)&Vt[dcol * 72 + kk + 16];
                v16h bf;
                #pragma unroll
                for (int i = 0; i < 8; ++i) { bf[i] = b0[i]; bf[8 + i] = b1[i]; }
                oacc[nt] = wmma16(ap[c], bf, oacc[nt]);
            }
        }
    }

    // write (B,S,D) output
    #pragma unroll
    for (int nt = 0; nt < 4; ++nt) {
        int d = nt * 16 + l15;
        #pragma unroll
        for (int j = 0; j < 8; ++j) {
            int rloc = wv * 16 + j + (hi ? 8 : 0);
            out[(rowQ0 + rloc) * D_ + colH + d] = oacc[nt][j] / lrow[j];
        }
    }
}

// -------------------------------------------- edge_mask: provably all-false -> 0
// scores = -sqrt(max(d2,1e-12)) <= -1e-6 < 0, so sigmoid(scores) > 0.5 never holds.
// Non-temporal: 256MB write-once stream, keep it out of L2.
__global__ __launch_bounds__(256) void fill_zero4_kernel(f32x4* __restrict__ p, size_t n4) {
    size_t i = (size_t)blockIdx.x * 256 + threadIdx.x;
    if (i < n4) {
        f32x4 z = {0.f, 0.f, 0.f, 0.f};
        __builtin_nontemporal_store(z, p + i);
    }
}

extern "C" void kernel_launch(void* const* d_in, const int* in_sizes, int n_in,
                              void* d_out, int out_size, void* d_ws, size_t ws_size,
                              hipStream_t stream) {
    (void)in_sizes; (void)n_in; (void)out_size; (void)ws_size;
    const float* x     = (const float*)d_in[0];
    const float* Wq    = (const float*)d_in[1];
    const float* bq    = (const float*)d_in[2];
    const float* Wk    = (const float*)d_in[3];
    const float* bk    = (const float*)d_in[4];
    const float* Wv    = (const float*)d_in[5];
    const float* bv    = (const float*)d_in[6];
    const float* gamma = (const float*)d_in[7];
    const float* beta  = (const float*)d_in[8];

    _Float16* xh   = (_Float16*)d_ws;                      // 8192*512
    _Float16* wh   = xh + (size_t)M_ * D_;                 // 3*512*512
    _Float16* qkvh = wh + (size_t)3 * D_ * D_;             // 3*8192*512

    cvt_f32_f16_kernel<<<(M_ * D_) / 1024, 256, 0, stream>>>(x, xh, M_ * D_);
    cvt_f32_f16_kernel<<<(D_ * D_) / 1024, 256, 0, stream>>>(Wq, wh, D_ * D_);
    cvt_f32_f16_kernel<<<(D_ * D_) / 1024, 256, 0, stream>>>(Wk, wh + (size_t)D_ * D_, D_ * D_);
    cvt_f32_f16_kernel<<<(D_ * D_) / 1024, 256, 0, stream>>>(Wv, wh + (size_t)2 * D_ * D_, D_ * D_);

    gemm_ln_kernel<<<dim3(M_ / 32, 1, 3), 256, 0, stream>>>(xh, wh, bq, bk, bv, gamma, beta, qkvh);

    attn_kernel<<<dim3(S_ / 64, H_, B_), 128, 0, stream>>>(qkvh, (float*)d_out);

    float* mask = (float*)d_out + (size_t)M_ * D_;
    size_t n4 = ((size_t)B_ * H_ * S_ * S_) / 4;           // 16,777,216 float4
    fill_zero4_kernel<<<(unsigned)((n4 + 255) / 256), 256, 0, stream>>>((f32x4*)mask, n4);
}
